// GCNNet_18915035972081
// MI455X (gfx1250) — compile-verified, matrix operand order checked by reading, not run on hardware
//
#include <hip/hip_runtime.h>
#include <hip/hip_bf16.h>

// ---------------------------------------------------------------------------
// Types for CDNA5 WMMA (wave32): bf16 A/B (16x32), f32 C/D (16x16, 8 VGPRs)
// ---------------------------------------------------------------------------
typedef __attribute__((ext_vector_type(16))) __bf16 v16bf;
typedef __attribute__((ext_vector_type(8)))  __bf16 v8bf;
typedef __attribute__((ext_vector_type(8)))  float  v8f;

static __device__ __forceinline__ __bf16 f2bf(float f) {
    unsigned u = __builtin_bit_cast(unsigned, f);
    u += 0x7FFFu + ((u >> 16) & 1u);            // round-to-nearest-even
    unsigned short s = (unsigned short)(u >> 16);
    return __builtin_bit_cast(__bf16, s);
}

static __device__ __forceinline__ v16bf load_frag(const __bf16* p0, const __bf16* p1) {
    v8bf lo = *(const v8bf*)p0;                 // 16B contiguous: global_load_b128
    v8bf hi = *(const v8bf*)p1;
    v16bf r;
#pragma unroll
    for (int i = 0; i < 8; ++i) { r[i] = lo[i]; r[i + 8] = hi[i]; }
    return r;
}

// ---------------------------------------------------------------------------
// WMMA GEMM: C[M,Nc] = A[M,K](bf16) * Bt[Nc,K](bf16)^T (+ bias)
// One wave per 16x16 tile. M%16==0, K%32==0, Nc%16==0 guaranteed by shapes.
// A-frag per 16-bit layout: lane<16 -> K = {k..k+7, k+16..k+23}; lane>=16 -> +8.
// ---------------------------------------------------------------------------
__global__ void gat_wmma_gemm(const __bf16* __restrict__ A,
                              const __bf16* __restrict__ Bt,
                              float* __restrict__ C,
                              const float* __restrict__ bias,
                              int M, int K, int Nc)
{
    int gt   = blockIdx.x * blockDim.x + threadIdx.x;
    int wave = gt >> 5;
    int lane = threadIdx.x & 31;
    int tilesM = M >> 4, tilesN = Nc >> 4;
    if (wave >= tilesM * tilesN) return;
    int tm = wave / tilesN, tn = wave % tilesN;
    int half = lane >> 4, r = lane & 15;

    const __bf16* arow = A  + (size_t)(tm * 16 + r) * K;
    const __bf16* brow = Bt + (size_t)(tn * 16 + r) * K;

    v8f c = {};
    for (int k = 0; k < K; k += 32) {
        int k0 = k + half * 8;
        int k1 = k + 16 + half * 8;
        v16bf a = load_frag(arow + k0, arow + k1);
        v16bf b = load_frag(brow + k0, brow + k1);
        c = __builtin_amdgcn_wmma_f32_16x16x32_bf16(
                false, a, false, b, (short)0, c, false, false);
    }

    int col = tn * 16 + r;
    float badd = bias ? bias[col] : 0.0f;
#pragma unroll
    for (int i = 0; i < 8; ++i) {
        int row = tm * 16 + i + 8 * half;
        C[(size_t)row * Nc + col] = c[i] + badd;
    }
}

// ---------------------------------------------------------------------------
// fp32 -> bf16 elementwise
// ---------------------------------------------------------------------------
__global__ void gat_cast_bf16(const float* __restrict__ in, __bf16* __restrict__ out,
                              long long n)
{
    long long i = (long long)blockIdx.x * blockDim.x + threadIdx.x;
    if (i < n) out[i] = f2bf(in[i]);
}

// W [Fin, Fout] fp32 -> Wt [Fout, Fin] bf16 (transpose so B frags are contiguous)
__global__ void gat_transpose_cast(const float* __restrict__ W, __bf16* __restrict__ Wt,
                                   int Fin, int Fout)
{
    int i = blockIdx.x * blockDim.x + threadIdx.x;
    if (i >= Fin * Fout) return;
    int fi = i / Fout, fo = i % Fout;
    Wt[(size_t)fo * Fin + fi] = f2bf(W[i]);
}

// ---------------------------------------------------------------------------
// Per-node prep: alpha_s/alpha_d dots, init max/sum, zero aggregation rows
// ---------------------------------------------------------------------------
__global__ void gat_node_prep(const float* __restrict__ h,
                              const float* __restrict__ a_s,
                              const float* __restrict__ a_d,
                              int N, int F,
                              float* __restrict__ as, float* __restrict__ ad,
                              float* __restrict__ nmax, float* __restrict__ nsum,
                              float* __restrict__ agg)
{
    int n = blockIdx.x * blockDim.x + threadIdx.x;
    if (n >= N) return;
    const float* row = h + (size_t)n * F;
    float s0 = 0.f, s1 = 0.f;
    for (int f = 0; f < F; ++f) { float v = row[f]; s0 += v * a_s[f]; s1 += v * a_d[f]; }
    as[n] = s0; ad[n] = s1;
    nmax[n] = -INFINITY; nsum[n] = 0.f;
    float* ar = agg + (size_t)n * F;
    for (int f = 0; f < F; ++f) ar[f] = 0.f;
}

// ---------------------------------------------------------------------------
// Edge softmax passes. Edges id<E come from edge_index; id>=E are self-loops.
// ---------------------------------------------------------------------------
static __device__ __forceinline__ void atomicMaxF32(float* addr, float val) {
    if (val >= 0.f) atomicMax((int*)addr, __builtin_bit_cast(int, val));
    else            atomicMin((unsigned int*)addr, __builtin_bit_cast(unsigned, val));
}

__global__ void gat_edge_pass1(const int* __restrict__ ei, int E, int N,
                               const float* __restrict__ as, const float* __restrict__ ad,
                               float* __restrict__ e_edge, float* __restrict__ nmax)
{
    int id = blockIdx.x * blockDim.x + threadIdx.x;
    if (id >= E + N) return;
    int s, d;
    if (id < E) { s = ei[id]; d = ei[E + id]; } else { s = d = id - E; }
    float v = as[s] + ad[d];
    v = (v > 0.f) ? v : 0.2f * v;               // leaky_relu(0.2)
    e_edge[id] = v;
    atomicMaxF32(&nmax[d], v);
}

__global__ void gat_edge_pass2(const int* __restrict__ ei, int E, int N,
                               float* __restrict__ e_edge,
                               const float* __restrict__ nmax, float* __restrict__ nsum)
{
    int id = blockIdx.x * blockDim.x + threadIdx.x;
    if (id >= E + N) return;
    int d = (id < E) ? ei[E + id] : id - E;
    float ex = __expf(e_edge[id] - nmax[d]);
    e_edge[id] = ex;                            // keep numerator for pass 3
    atomicAdd(&nsum[d], ex);
}

// Weighted scatter-add: 32 consecutive threads cover one edge's feature row
// (coalesced 512B gather at F=128); atomics land in L2 (agg fits easily).
__global__ void gat_edge_aggregate(const int* __restrict__ ei, int E, int N, int F,
                                   const float* __restrict__ e_edge,
                                   const float* __restrict__ nsum,
                                   const float* __restrict__ h,
                                   float* __restrict__ agg)
{
    long long t = (long long)blockIdx.x * blockDim.x + threadIdx.x;
    int chunks = F >> 2;
    if (t >= (long long)(E + N) * chunks) return;
    int id = (int)(t / chunks);
    int c  = (int)(t % chunks);
    int s, d;
    if (id < E) { s = ei[id]; d = ei[E + id]; } else { s = d = id - E; }
    float w = e_edge[id] / nsum[d];
    float4 hv = ((const float4*)(h + (size_t)s * F))[c];
    float* out = agg + (size_t)d * F + (size_t)c * 4;
    atomicAdd(out + 0, hv.x * w);
    atomicAdd(out + 1, hv.y * w);
    atomicAdd(out + 2, hv.z * w);
    atomicAdd(out + 3, hv.w * w);
}

// out = relu(agg + b) -> bf16 activations for next layer's WMMA GEMM
__global__ void gat_bias_relu_cast(const float* __restrict__ agg,
                                   const float* __restrict__ b,
                                   __bf16* __restrict__ xb, int N, int F)
{
    long long i = (long long)blockIdx.x * blockDim.x + threadIdx.x;
    if (i >= (long long)N * F) return;
    float v = agg[i] + b[i % F];
    xb[i] = f2bf(v > 0.f ? v : 0.f);
}

// ---------------------------------------------------------------------------
// Host launch
// ---------------------------------------------------------------------------
static inline int cdiv(long long a, long long b) { return (int)((a + b - 1) / b); }

extern "C" void kernel_launch(void* const* d_in, const int* in_sizes, int n_in,
                              void* d_out, int out_size, void* d_ws, size_t ws_size,
                              hipStream_t stream)
{
    const int INP = 128;
    const int N = in_sizes[0] / INP;
    const int E = in_sizes[1] / 2;
    const int dims[5] = {128, 32, 64, 128, 128};

    const float* x  = (const float*)d_in[0];
    const int*   ei = (const int*)d_in[1];

    // workspace carve-up
    char* w = (char*)d_ws;
    __bf16* xb   = (__bf16*)w;  w += (size_t)N * 128 * sizeof(__bf16);
    float*  h    = (float*)w;   w += (size_t)N * 128 * sizeof(float);
    float*  agg  = (float*)w;   w += (size_t)N * 128 * sizeof(float);
    float*  as   = (float*)w;   w += (size_t)N * sizeof(float);
    float*  ad   = (float*)w;   w += (size_t)N * sizeof(float);
    float*  nmax = (float*)w;   w += (size_t)N * sizeof(float);
    float*  nsum = (float*)w;   w += (size_t)N * sizeof(float);
    float*  ee   = (float*)w;   w += (size_t)(E + N) * sizeof(float);
    __bf16* Wt   = (__bf16*)w;  // <= 128*128 bf16

    const int B = 256;

    // x -> bf16
    gat_cast_bf16<<<cdiv((long long)N * INP, B), B, 0, stream>>>(x, xb, (long long)N * INP);

    for (int layer = 0; layer < 4; ++layer) {
        const int Fin  = dims[layer];
        const int Fout = dims[layer + 1];
        const float* Wl  = (const float*)d_in[2 + 4 * layer + 0];
        const float* asl = (const float*)d_in[2 + 4 * layer + 1];
        const float* adl = (const float*)d_in[2 + 4 * layer + 2];
        const float* bl  = (const float*)d_in[2 + 4 * layer + 3];

        gat_transpose_cast<<<cdiv(Fin * Fout, B), B, 0, stream>>>(Wl, Wt, Fin, Fout);

        long long waves = (long long)(N / 16) * (Fout / 16);
        gat_wmma_gemm<<<cdiv(waves * 32, B), B, 0, stream>>>(xb, Wt, h, nullptr, N, Fin, Fout);

        gat_node_prep<<<cdiv(N, B), B, 0, stream>>>(h, asl, adl, N, Fout, as, ad, nmax, nsum, agg);

        gat_edge_pass1<<<cdiv(E + N, B), B, 0, stream>>>(ei, E, N, as, ad, ee, nmax);
        gat_edge_pass2<<<cdiv(E + N, B), B, 0, stream>>>(ei, E, N, ee, nmax, nsum);
        gat_edge_aggregate<<<cdiv((long long)(E + N) * (Fout / 4), B), B, 0, stream>>>(
            ei, E, N, Fout, ee, nsum, h, agg);

        gat_bias_relu_cast<<<cdiv((long long)N * Fout, B), B, 0, stream>>>(agg, bl, xb, N, Fout);
    }

    // FC head: [N,128] @ [128,16] + b -> d_out (fp32)
    const float* fcW = (const float*)d_in[18];
    const float* fcb = (const float*)d_in[19];
    gat_transpose_cast<<<cdiv(128 * 16, B), B, 0, stream>>>(fcW, Wt, 128, 16);
    long long waves = (long long)(N / 16) * (16 / 16);
    gat_wmma_gemm<<<cdiv(waves * 32, B), B, 0, stream>>>(xb, Wt, (float*)d_out, fcb, N, 128, 16);
}
